// AttentiveRNNLanguageModel_31267361915371
// MI455X (gfx1250) — matrix operand order repaired and checked by reading, not compiled, not run on hardware
//
#include <hip/hip_runtime.h>
#include <hip/hip_bf16.h>
#include <math.h>

// ---------------------------------------------------------------------------
// AttentiveRNNLanguageModel for MI455X (gfx1250, wave32, WMMA bf16 + TDM)
// ---------------------------------------------------------------------------
// Shapes: V=32000, E=H=1024, B=8, T=512, BT=4096.
//   * All big GEMMs: bf16 WMMA (v_wmma_f32_16x16x32_bf16), f32 accumulate.
//   * GEMM tiles moved by the Tensor Data Mover (tensor_load_to_lds),
//     double-buffered LDS, TENSORcnt-synchronized (CDNA5-idiomatic path).
//   * Decoder GEMM (4096x32000x1024, 268 GFLOP) dominates; emb_bf16 (65.5MB)
//     stays L2-resident (192MB L2).
//   * LSTM recurrence: serial -> 512 per-step kernels; Whh L2-resident.
//   * Causal-prefix softmax == running (max,sum,acc) scan: O(T*H).
// ---------------------------------------------------------------------------

typedef __attribute__((ext_vector_type(16))) __bf16 v16bf;
typedef __attribute__((ext_vector_type(8)))  __bf16 v8bf;
typedef __attribute__((ext_vector_type(8)))  float  v8f;
typedef unsigned int u32x4 __attribute__((ext_vector_type(4)));
typedef int          i32x8 __attribute__((ext_vector_type(8)));
typedef int          i32x4 __attribute__((ext_vector_type(4)));

#define LM_B 8
#define LM_T 512
#define LM_H 1024
#define LM_E 1024
#define LM_V 32000

// ---------------------------- utility kernels ------------------------------

__global__ void k_zero_f32(float* p, int n) {
  int i = blockIdx.x * blockDim.x + threadIdx.x;
  if (i < n) p[i] = 0.f;
}

__global__ void k_add_f32(const float* __restrict__ a, const float* __restrict__ b,
                          float* __restrict__ o, int n) {
  int i = blockIdx.x * blockDim.x + threadIdx.x;
  if (i < n) o[i] = a[i] + b[i];
}

__global__ void k_cvt_bf16(const float* __restrict__ in, __bf16* __restrict__ out, size_t n) {
  size_t i = (size_t)blockIdx.x * blockDim.x + threadIdx.x;
  size_t stride = (size_t)gridDim.x * blockDim.x;
  for (; i < n; i += stride) out[i] = (__bf16)in[i];
}

// x_bf16[m][e] = bf16(emb[ids[m]][e]) ; one block per token position
__global__ void k_gather_bf16(const int* __restrict__ ids, const float* __restrict__ emb,
                              __bf16* __restrict__ out, int E) {
  int m = blockIdx.x;
  const float* src = emb + (size_t)ids[m] * E;
  __bf16* dst = out + (size_t)m * E;
  for (int e = threadIdx.x; e < E; e += blockDim.x) dst[e] = (__bf16)src[e];
}

// ------------------------------ WMMA GEMM ----------------------------------
// C[M][N] = A[M][K] * B[N][K]^T + bias[N]  (optional tanh, f32 or bf16 out)
// Block: 256 threads = 8 wave32, 128x128 tile, per-wave 64x32 (4x2 WMMA tiles)
// Tiles staged into LDS by the Tensor Data Mover, double-buffered.

#define BM 128
#define BN 128
#define BK 32
#define LPAD 8
#define LSTR (BK + LPAD)          // 40 bf16 = 80B row stride (TDM pads per row)
#define TILE_ELEMS (BM * LSTR)    // 5120 bf16 = 10240 B per tile
#define STAGE_ELEMS (2 * TILE_ELEMS)

union V16U { v16bf v; v8bf h[2]; };

// Issue one 2D TDM tile load: tile0 x tile1 elements of 2B data, row stride
// `stride0` elements, into LDS at byte offset lds_off. Rows are padded in LDS:
// pad_interval code 3 (16 DWORDs = 64B = one 32-elem bf16 row), pad_amount
// code 3 (4 DWORDs = 16B = 8 bf16)  =>  LDS row stride 80B == LSTR.
__device__ __forceinline__ void tdm_load_tile_bf16(
    unsigned lds_off, const __bf16* gptr,
    unsigned tensor_d0, unsigned tensor_d1, unsigned stride0,
    unsigned tile0, unsigned tile1)
{
  unsigned long long ga = (unsigned long long)(size_t)gptr;
  u32x4 g0;
  g0[0] = 1u;                                           // count=1 (valid D#)
  g0[1] = lds_off;                                      // lds_addr [63:32]
  g0[2] = (unsigned)(ga & 0xFFFFFFFFu);                 // global_addr lo
  g0[3] = (unsigned)((ga >> 32) & 0x01FFFFFFu) | (2u << 30); // ga hi | type=2

  i32x8 g1;
  g1[0] = (int)((1u << 16)      // data_size = 1 -> 2 bytes
              | (1u << 20)      // pad_enable
              | (3u << 22)      // pad_interval: 16 DWORDs
              | (3u << 25));    // pad_amount: 4 DWORDs
  g1[1] = (int)((tensor_d0 & 0xFFFFu) << 16);                         // td0 lo
  g1[2] = (int)(((tensor_d0 >> 16) & 0xFFFFu) | ((tensor_d1 & 0xFFFFu) << 16));
  g1[3] = (int)(((tensor_d1 >> 16) & 0xFFFFu) | (tile0 << 16));       // tile0
  g1[4] = (int)tile1;                                   // tile1 | tile2=0
  g1[5] = (int)stride0;                                 // dim0 stride lo32
  g1[6] = 0;                                            // stride hi | d1s lo
  g1[7] = 0;
  i32x4 z4 = {0, 0, 0, 0};                              // groups 2,3: 2D only
  i32x8 z8 = {0, 0, 0, 0, 0, 0, 0, 0};                  // extra group (clang-23 form)
  __builtin_amdgcn_tensor_load_to_lds(g0, g1, z4, z4, z8, 0);
}

template <bool TANH, bool OUT_BF16>
__global__ __launch_bounds__(256) void k_gemm_bf16(
    const __bf16* __restrict__ A,   // [M][K]
    const __bf16* __restrict__ Bm,  // [N][K]
    const float*  __restrict__ bias,// [N] or null
    float*  __restrict__ outF,
    __bf16* __restrict__ outB,
    int M, int N, int K)
{
  __shared__ __align__(16) __bf16 lds[2 * STAGE_ELEMS];  // 40 KB

  const int tid  = threadIdx.x;
  const int lane = tid & 31;
  const int wave = tid >> 5;
  const int bm = blockIdx.y * BM;
  const int bn = blockIdx.x * BN;
  const int wm = (wave >> 2) * 64;   // 0 | 64
  const int wn = (wave & 3) * 32;    // 0,32,64,96

  // raw LDS byte address: low 32 bits of the generic pointer (aperture|offset)
  const unsigned ldsBase = (unsigned)(size_t)(void*)&lds[0];

  // fragment lane mapping (ISA 7.12.2)
  const int frow  = lane & 15;
  const int ahalf = (lane < 16) ? 0 : 8;    // A: K slices {a..a+7, a+16..a+23}
  const int bhalf = (lane < 16) ? 0 : 8;    // B: K slices {b..b+7, b+16..b+23}

  v8f acc[4][2] = {};

  const int nk = K / BK;

  // prologue: stage 0
  if (wave == 0) {
    tdm_load_tile_bf16(ldsBase,                     A  + (size_t)bm * K,
                       (unsigned)K, (unsigned)M, (unsigned)K, BK, BM);
    tdm_load_tile_bf16(ldsBase + TILE_ELEMS * 2,    Bm + (size_t)bn * K,
                       (unsigned)K, (unsigned)N, (unsigned)K, BK, BN);
  }

  for (int i = 0; i < nk; ++i) {
    const int cur = i & 1;
    const int nxt = cur ^ 1;
    if (wave == 0) {
      if (i + 1 < nk) {
        const int kk = (i + 1) * BK;
        const unsigned so = (unsigned)(nxt * STAGE_ELEMS * 2);
        tdm_load_tile_bf16(ldsBase + so,                  A  + (size_t)bm * K + kk,
                           (unsigned)K, (unsigned)M, (unsigned)K, BK, BM);
        tdm_load_tile_bf16(ldsBase + so + TILE_ELEMS * 2, Bm + (size_t)bn * K + kk,
                           (unsigned)K, (unsigned)N, (unsigned)K, BK, BN);
        __builtin_amdgcn_s_wait_tensorcnt((short)2);  // retire current stage
      } else {
        __builtin_amdgcn_s_wait_tensorcnt((short)0);
      }
    }
    __syncthreads();  // current stage visible to all waves

    const __bf16* sA = lds + cur * STAGE_ELEMS;
    const __bf16* sB = sA + TILE_ELEMS;

    v16bf afrag[4], bfrag[2];
#pragma unroll
    for (int mt = 0; mt < 4; ++mt) {
      const __bf16* p = &sA[(wm + mt * 16 + frow) * LSTR + ahalf];
      V16U u; u.h[0] = *(const v8bf*)p; u.h[1] = *(const v8bf*)(p + 16);
      afrag[mt] = u.v;
    }
#pragma unroll
    for (int nt = 0; nt < 2; ++nt) {
      const __bf16* p = &sB[(wn + nt * 16 + frow) * LSTR + bhalf];
      V16U u; u.h[0] = *(const v8bf*)p; u.h[1] = *(const v8bf*)(p + 16);
      bfrag[nt] = u.v;
    }
#pragma unroll
    for (int mt = 0; mt < 4; ++mt)
#pragma unroll
      for (int nt = 0; nt < 2; ++nt)
        acc[mt][nt] = __builtin_amdgcn_wmma_f32_16x16x32_bf16(
            false, afrag[mt], false, bfrag[nt], (short)0, acc[mt][nt], false, false);

    __syncthreads();  // all reads done before stage is overwritten
  }

  // epilogue: D VGPR v -> row v + (lane>=16 ? 8 : 0), col = lane&15
  const int crow = (lane < 16) ? 0 : 8;
#pragma unroll
  for (int mt = 0; mt < 4; ++mt) {
#pragma unroll
    for (int nt = 0; nt < 2; ++nt) {
      const int gcol = bn + wn + nt * 16 + frow;
      const float bv = bias ? bias[gcol] : 0.f;
#pragma unroll
      for (int v = 0; v < 8; ++v) {
        const int grow = bm + wm + mt * 16 + crow + v;
        float val = acc[mt][nt][v] + bv;
        if (TANH) val = tanhf(val);
        if (OUT_BF16) outB[(size_t)grow * N + gcol] = (__bf16)val;
        else          outF[(size_t)grow * N + gcol] = val;
      }
    }
  }
}

// ------------------------------ LSTM step ----------------------------------
// gates = x_pre[:,t,:] + h_prev @ Whh^T ; pointwise update; serial over t.
// grid = B * (H/256) = 32 blocks, thread -> one (b, j)

__device__ __forceinline__ float sigmoidf_(float x) { return 1.f / (1.f + __expf(-x)); }

__global__ __launch_bounds__(256) void k_lstm_step(
    const float* __restrict__ x_pre,  // [B*T][4H]
    const float* __restrict__ Whh,    // [4H][H]
    const float* __restrict__ h_in, const float* __restrict__ c_in,   // [B][H]
    float* __restrict__ h_out, float* __restrict__ c_out,
    float* __restrict__ enc,          // [B*T][H]
    float* __restrict__ hT, float* __restrict__ cT,  // null except t==T-1
    int t, int T, int H)
{
  __shared__ float hs[LM_H];
  const int b = blockIdx.x >> 2;
  const int j = (blockIdx.x & 3) * 256 + threadIdx.x;
  for (int k = threadIdx.x; k < H; k += 256) hs[k] = h_in[b * H + k];
  __syncthreads();

  const float* xp = x_pre + (size_t)(b * T + t) * (4 * H);
  float dots[4];
#pragma unroll
  for (int g = 0; g < 4; ++g) {
    const float4* wr = (const float4*)(Whh + (size_t)(g * H + j) * H);
    float a = 0.f;
    for (int k4 = 0; k4 < H / 4; ++k4) {
      float4 w = wr[k4];
      float4 hv = *(const float4*)&hs[k4 * 4];
      a += w.x * hv.x + w.y * hv.y + w.z * hv.z + w.w * hv.w;
    }
    dots[g] = a;
  }
  const float ig = sigmoidf_(xp[j]         + dots[0]);
  const float fg = sigmoidf_(xp[H + j]     + dots[1]);
  const float gg = tanhf    (xp[2 * H + j] + dots[2]);
  const float og = sigmoidf_(xp[3 * H + j] + dots[3]);
  const float c  = fg * c_in[b * H + j] + ig * gg;
  const float h  = og * tanhf(c);
  c_out[b * H + j] = c;
  h_out[b * H + j] = h;
  enc[(size_t)(b * T + t) * H + j] = h;
  if (hT) { hT[b * H + j] = h; cT[b * H + j] = c; }
}

// --------------------------- attention scores ------------------------------
// s[m] = sum_k z_bf16[m][k] * W2[k] + b2 ; one block per row

__global__ __launch_bounds__(256) void k_score(
    const __bf16* __restrict__ z, const float* __restrict__ W2,
    const float* __restrict__ b2, float* __restrict__ s, int H)
{
  __shared__ float red[256];
  const int m = blockIdx.x;
  float a = 0.f;
  for (int k = threadIdx.x; k < H; k += 256)
    a += (float)z[(size_t)m * H + k] * W2[k];
  red[threadIdx.x] = a;
  __syncthreads();
  for (int off = 128; off > 0; off >>= 1) {
    if (threadIdx.x < off) red[threadIdx.x] += red[threadIdx.x + off];
    __syncthreads();
  }
  if (threadIdx.x == 0) s[m] = red[0] + b2[0];
}

// ------------------- causal-prefix softmax + context scan ------------------
// ctx[b,t,:] = softmax(s[b,0..t]) @ enc[b,0..t,:]  -> running-max/sum scan.
// Writes cat[(b*T+t)][0..H) = bf16(ctx), cat[..][H..2H) = bf16(enc).

__global__ __launch_bounds__(256) void k_attn_scan(
    const float* __restrict__ s,    // [B*T]
    const float* __restrict__ enc,  // [B*T][H]
    __bf16* __restrict__ cat,       // [B*T][2H]
    int T, int H)
{
  const int b = blockIdx.x >> 2;
  const int d = (blockIdx.x & 3) * 256 + threadIdx.x;
  float m = -1e30f, Z = 0.f, acc = 0.f;
  for (int t = 0; t < T; ++t) {
    const float sv = s[b * T + t];
    const float mn = fmaxf(m, sv);
    const float scale = __expf(m - mn);
    const float e = __expf(sv - mn);
    const float ev = enc[(size_t)(b * T + t) * H + d];
    acc = acc * scale + e * ev;
    Z   = Z   * scale + e;
    m = mn;
    const size_t row = (size_t)(b * T + t) * (2 * H);
    cat[row + d]     = (__bf16)(acc / Z);
    cat[row + H + d] = (__bf16)ev;
  }
}

// ------------------------------- launcher ----------------------------------

extern "C" void kernel_launch(void* const* d_in, const int* in_sizes, int n_in,
                              void* d_out, int out_size, void* d_ws, size_t ws_size,
                              hipStream_t stream) {
  (void)in_sizes; (void)n_in; (void)out_size; (void)ws_size;

  const int*   ids   = (const int*)  d_in[0];
  // d_in[1]=h0, d_in[2]=c0 unused (reference starts from zeros)
  const float* emb   = (const float*)d_in[3];
  const float* Wih   = (const float*)d_in[4];
  const float* Whh   = (const float*)d_in[5];
  const float* bih   = (const float*)d_in[6];
  const float* bhh   = (const float*)d_in[7];
  const float* W1    = (const float*)d_in[8];
  const float* b1    = (const float*)d_in[9];
  const float* W2    = (const float*)d_in[10];
  const float* b2    = (const float*)d_in[11];
  const float* Wc    = (const float*)d_in[12];
  const float* bc    = (const float*)d_in[13];
  const float* dec_b = (const float*)d_in[14];

  const int BT = LM_B * LM_T;  // 4096
  char* ws = (char*)d_ws;
  size_t off = 0;
  auto alloc = [&](size_t bytes) -> char* {
    char* p = ws + off;
    off = (off + bytes + 255) & ~(size_t)255;
    return p;
  };

  __bf16* wih_b = (__bf16*)alloc((size_t)4 * LM_H * LM_E * 2);
  __bf16* w1_b  = (__bf16*)alloc((size_t)LM_H * LM_H * 2);
  __bf16* wc_b  = (__bf16*)alloc((size_t)LM_H * 2 * LM_H * 2);
  __bf16* emb_b = (__bf16*)alloc((size_t)LM_V * LM_E * 2);
  float*  bsum  = (float*) alloc((size_t)4 * LM_H * 4);
  __bf16* x_b   = (__bf16*)alloc((size_t)BT * LM_E * 2);
  float*  xpre  = (float*) alloc((size_t)BT * 4 * LM_H * 4);
  float*  hbuf0 = (float*) alloc((size_t)LM_B * LM_H * 4);
  float*  hbuf1 = (float*) alloc((size_t)LM_B * LM_H * 4);
  float*  cbuf0 = (float*) alloc((size_t)LM_B * LM_H * 4);
  float*  cbuf1 = (float*) alloc((size_t)LM_B * LM_H * 4);
  float*  enc   = (float*) alloc((size_t)BT * LM_H * 4);
  __bf16* enc_b = (__bf16*)alloc((size_t)BT * LM_H * 2);
  __bf16* z_b   = (__bf16*)alloc((size_t)BT * LM_H * 2);
  float*  sarr  = (float*) alloc((size_t)BT * 4);
  __bf16* cat_b = (__bf16*)alloc((size_t)BT * 2 * LM_H * 2);
  __bf16* comb  = (__bf16*)alloc((size_t)BT * LM_H * 2);

  float* dec = (float*)d_out;
  float* hT  = dec + (size_t)BT * LM_V;
  float* cT  = hT + (size_t)LM_B * LM_H;

  // weight conversions (fp32 -> bf16)
  k_cvt_bf16<<<1024, 256, 0, stream>>>(Wih, wih_b, (size_t)4 * LM_H * LM_E);
  k_cvt_bf16<<<512,  256, 0, stream>>>(W1,  w1_b,  (size_t)LM_H * LM_H);
  k_cvt_bf16<<<512,  256, 0, stream>>>(Wc,  wc_b,  (size_t)LM_H * 2 * LM_H);
  k_cvt_bf16<<<2048, 256, 0, stream>>>(emb, emb_b, (size_t)LM_V * LM_E);
  k_add_f32<<<(4 * LM_H) / 256, 256, 0, stream>>>(bih, bhh, bsum, 4 * LM_H);

  // embedding gather (bf16)
  k_gather_bf16<<<BT, 256, 0, stream>>>(ids, emb, x_b, LM_E);

  // x_pre = x @ Wih^T + (bih+bhh) : M=4096 N=4096 K=1024, f32 out
  k_gemm_bf16<false, false><<<dim3(4 * LM_H / BN, BT / BM), 256, 0, stream>>>(
      x_b, wih_b, bsum, xpre, nullptr, BT, 4 * LM_H, LM_E);

  // LSTM recurrence (serial over T)
  k_zero_f32<<<(LM_B * LM_H) / 256, 256, 0, stream>>>(hbuf0, LM_B * LM_H);
  k_zero_f32<<<(LM_B * LM_H) / 256, 256, 0, stream>>>(cbuf0, LM_B * LM_H);
  for (int t = 0; t < LM_T; ++t) {
    float* hin  = (t & 1) ? hbuf1 : hbuf0;
    float* hout = (t & 1) ? hbuf0 : hbuf1;
    float* cin  = (t & 1) ? cbuf1 : cbuf0;
    float* cout = (t & 1) ? cbuf0 : cbuf1;
    const bool last = (t == LM_T - 1);
    k_lstm_step<<<32, 256, 0, stream>>>(xpre, Whh, hin, cin, hout, cout, enc,
                                        last ? hT : nullptr, last ? cT : nullptr,
                                        t, LM_T, LM_H);
  }

  // z = tanh(enc @ W1^T + b1) : bf16 out
  k_cvt_bf16<<<2048, 256, 0, stream>>>(enc, enc_b, (size_t)BT * LM_H);
  k_gemm_bf16<true, true><<<dim3(LM_H / BN, BT / BM), 256, 0, stream>>>(
      enc_b, w1_b, b1, nullptr, z_b, BT, LM_H, LM_H);

  // s = z @ W2 + b2
  k_score<<<BT, 256, 0, stream>>>(z_b, W2, b2, sarr, LM_H);

  // causal softmax scan -> cat = [ctx | enc] in bf16
  k_attn_scan<<<LM_B * (LM_H / 256), 256, 0, stream>>>(sarr, enc, cat_b, LM_T, LM_H);

  // combined = tanh(cat @ Wc^T + bc) : bf16 out, K=2048
  k_gemm_bf16<true, true><<<dim3(LM_H / BN, BT / BM), 256, 0, stream>>>(
      cat_b, wc_b, bc, nullptr, comb, BT, LM_H, 2 * LM_H);

  // decoded = combined @ emb^T + dec_b : M=4096 N=32000 K=1024, f32 out
  k_gemm_bf16<false, false><<<dim3(LM_V / BN, BT / BM), 256, 0, stream>>>(
      comb, emb_b, dec_b, dec, nullptr, BT, LM_V, LM_E);
}
